// SDGraphUNet_7945689498143
// MI455X (gfx1250) — matrix operand order repaired and checked by reading, not compiled, not run on hardware
//
#include <hip/hip_runtime.h>
#include <stdint.h>

typedef _Float16 f16;
typedef __attribute__((ext_vector_type(16))) _Float16 v16h;
typedef __attribute__((ext_vector_type(8)))  _Float16 v8h;
typedef __attribute__((ext_vector_type(8)))  float    v8f;

#define N_PNT  4096
#define N_STK  32
#define N_SPNT 128
#define CHU    256   // union feature dim

// ---------------- workspace layout (bytes) ----------------
constexpr size_t OFF_XH    = 0;                                  // f16 [4][4096][256]
constexpr size_t OFF_SQ    = OFF_XH    + 4ull*4096*256*2;        // f32 [4][4096]
constexpr size_t OFF_US    = OFF_SQ    + 4ull*4096*4;            // f32 [4][256][32] union_sparse
constexpr size_t OFF_SUP   = OFF_US    + 4ull*256*32*4;          // f32 [4][256][32] sparse_up
constexpr size_t OFF_FPS   = OFF_SUP   + 4ull*256*32*4;          // int [4][16]
constexpr size_t OFF_SKNN  = OFF_FPS   + 1024;                   // int [4][32][2]
constexpr size_t OFF_IDX10 = OFF_SKNN  + 1024;                   // int [4][4096][12]
constexpr size_t OFF_DUP   = OFF_IDX10 + 4ull*4096*12*4;         // f32 [4][256][4096] dense_up
constexpr size_t OFF_WD2S  = OFF_DUP   + 4ull*256*4096*4;        // f16 [128][384]
constexpr size_t OFF_WDS   = OFF_WD2S  + 49152ull*2;             // f16 [256][768]
constexpr size_t OFF_WDN1  = OFF_WDS   + 196608ull*2;            // f16 [256][256]  (Wtop-Wbot)^T
constexpr size_t OFF_WDN2  = OFF_WDN1  + 65536ull*2;             // f16 [256][256]  Wbot^T
constexpr size_t OFF_BIG0  = OFF_WDN2  + 65536ull*2;             // im2col_d2s (12.4MB) then U (16.8MB)
constexpr size_t OFF_BIG1  = OFF_BIG0  + 4ull*4096*256*4;        // conv_d2s (8.3MB) then im2col_ds (6.3MB)
constexpr size_t OFF_BIG2  = OFF_BIG1  + 4ull*128*4032*4;        // V (16.8MB)

constexpr long long OUT_DN   = 4LL*256*16;                       // 16384
constexpr long long OUT_COOR = OUT_DN + 4LL*256*1024;            // 1064960

// ---------------- helpers ----------------
__device__ __forceinline__ void ins10(float* bv, int* bi, float v, int i) {
  if (v > bv[9] || (v == bv[9] && i >= bi[9])) return;
  int p = 9;
  while (p > 0 && (bv[p-1] > v || (bv[p-1] == v && bi[p-1] > i))) {
    bv[p] = bv[p-1]; bi[p] = bi[p-1]; --p;
  }
  bv[p] = v; bi[p] = i;
}

// ---------------- generic WMMA GEMM: C[M][N] = A[M][K] * Bt[N][K]^T ----------------
// grid: (N/64, M/16, batch); block: 128 (4 waves, one 16x16 C tile each)
// fuse=1: C = relu(g[m]*(acc + bias[m]) + bt[m])
__global__ void __launch_bounds__(128)
gemm_f16(const f16* __restrict__ A, const f16* __restrict__ Bt, float* __restrict__ C,
         int K, int N, long long sA, long long sB, long long sC,
         const float* __restrict__ g, const float* __restrict__ bias,
         const float* __restrict__ bt, int fuse) {
  int lane = threadIdx.x & 31, wv = threadIdx.x >> 5;
  int b  = blockIdx.z;
  int n0 = (blockIdx.x * 4 + wv) * 16;
  int m0 = blockIdx.y * 16;
  int h = lane >> 4, l15 = lane & 15;
  const f16* Ar = A + (long long)b * sA + (long long)(m0 + l15) * K;
  const f16* Br = Bt + (long long)b * sB + (long long)(n0 + l15) * K;
  v8f acc = {};
  for (int k0 = 0; k0 < K; k0 += 32) {
    union { v16h v; v8h p[2]; } a;
    a.p[0] = *(const v8h*)(Ar + k0 + h * 8);
    a.p[1] = *(const v8h*)(Ar + k0 + 16 + h * 8);
    union { v16h v; v8h p[2]; } bu;
    bu.p[0] = *(const v8h*)(Br + k0 + h * 8);
    bu.p[1] = *(const v8h*)(Br + k0 + 16 + h * 8);
    v16h bf = bu.v;
    acc = __builtin_amdgcn_wmma_f32_16x16x32_f16(false, a.v, false, bf, (short)0, acc,
                                                 false, false);
  }
  float* Cr = C + (long long)b * sC;
  int n = n0 + l15;
#pragma unroll
  for (int r = 0; r < 8; ++r) {
    int m = m0 + h * 8 + r;
    float v = acc[r];
    if (fuse) { v = g[m] * (v + bias[m]) + bt[m]; v = v > 0.f ? v : 0.f; }
    Cr[(long long)m * N + n] = v;
  }
}

// ---------------- Xh (f16 transposed union_dense) + squared norms ----------------
__global__ void build_xh(const float* __restrict__ dense, const float* __restrict__ sparse,
                         f16* __restrict__ Xh, float* __restrict__ sq) {
  int b = blockIdx.y;
  int n = blockIdx.x * 256 + threadIdx.x;
  const float* dp = dense + (long long)b * 128 * N_PNT;
  const float* sp = sparse + (long long)b * 128 * N_STK;
  f16* xp = Xh + ((long long)b * N_PNT + n) * CHU;
  int s = n >> 7;
  float acc = 0.f;
  for (int c = 0; c < 128; ++c) { float v = dp[c * N_PNT + n]; xp[c] = (f16)v; acc += v * v; }
  for (int c = 0; c < 128; ++c) { float v = sp[c * N_STK + s]; xp[128 + c] = (f16)v; acc += v * v; }
  sq[b * N_PNT + n] = acc;
}

// ---------------- im2col for d2s conv: Bt[(s*126+w)][i*3+dw] ----------------
__global__ void im2col_d2s_k(const float* __restrict__ dense, f16* __restrict__ B) {
  int s = blockIdx.x, b = blockIdx.y, i = threadIdx.x;
  const float* dp = dense + ((long long)b * 128 + i) * N_PNT + s * N_SPNT;
  f16* bp = B + (long long)b * 4032 * 384 + (long long)s * 126 * 384 + i * 3;
  for (int w = 0; w < 126; ++w) {
#pragma unroll
    for (int dw = 0; dw < 3; ++dw) bp[w * 384 + dw] = (f16)dp[w + dw];
  }
}

// ---------------- max over w + union_sparse assembly ----------------
__global__ void reduce_union(const float* __restrict__ conv, const float* __restrict__ sparse,
                             float* __restrict__ us) {
  int s = blockIdx.x, b = blockIdx.y, c = threadIdx.x;
  const float* cp = conv + ((long long)b * 128 + c) * 4032 + s * 126;
  float m = -3e38f;
  for (int w = 0; w < 126; ++w) m = fmaxf(m, cp[w]);
  us[((long long)b * 256 + 128 + c) * N_STK + s] = m;
  us[((long long)b * 256 + c) * N_STK + s] = sparse[((long long)b * 128 + c) * N_STK + s];
}

// ---------------- fused WMMA Gram + streaming top-10 kNN ----------------
__global__ void __launch_bounds__(256)
knn_dense(const f16* __restrict__ Xh, const float* __restrict__ sq, int* __restrict__ idx10) {
  __shared__ float distbuf[16][516];
  __shared__ float sqA[16];
  int tid = threadIdx.x, lane = tid & 31, wv = tid >> 5;
  int b = blockIdx.y, n0 = blockIdx.x * 16;
  const f16* Xb = Xh + (long long)b * N_PNT * CHU;
  const float* sqb = sq + (long long)b * N_PNT;
  if (tid < 16) sqA[tid] = sqb[n0 + tid];
  __syncthreads();
  float bv[10]; int bi[10];
#pragma unroll
  for (int q = 0; q < 10; ++q) { bv[q] = 3.0e38f; bi[q] = 0x7fffffff; }
  int h = lane >> 4, l15 = lane & 15;
  const f16* Ar = Xb + (long long)(n0 + l15) * CHU;
  for (int chunk = 0; chunk < 8; ++chunk) {
    int mbase = chunk * 512;
    for (int t = 0; t < 4; ++t) {
      int mt = mbase + (wv * 4 + t) * 16;
      const f16* Br = Xb + (long long)(mt + l15) * CHU;
      v8f acc = {};
      for (int k0 = 0; k0 < CHU; k0 += 32) {
        union { v16h v; v8h p[2]; } a;
        a.p[0] = *(const v8h*)(Ar + k0 + h * 8);
        a.p[1] = *(const v8h*)(Ar + k0 + 16 + h * 8);
        union { v16h v; v8h p[2]; } bu;
        bu.p[0] = *(const v8h*)(Br + k0 + h * 8);
        bu.p[1] = *(const v8h*)(Br + k0 + 16 + h * 8);
        v16h bf = bu.v;
        acc = __builtin_amdgcn_wmma_f32_16x16x32_f16(false, a.v, false, bf, (short)0, acc,
                                                     false, false);
      }
      float sqB = sqb[mt + l15];
      int colc = (wv * 4 + t) * 16 + l15;
#pragma unroll
      for (int r = 0; r < 8; ++r) {
        int i = h * 8 + r;
        distbuf[i][colc] = sqA[i] + sqB - 2.0f * acc[r];
      }
    }
    __syncthreads();
    if (tid < 128) {
      int row = tid >> 3, slot = tid & 7;
      for (int c = slot * 64; c < slot * 64 + 64; ++c)
        ins10(bv, bi, distbuf[row][c], mbase + c);
    }
    __syncthreads();
  }
  float* cval = &distbuf[0][0];
  int*   cidx = (int*)(cval + 16 * 80);
  if (tid < 128) {
    int row = tid >> 3, slot = tid & 7;
#pragma unroll
    for (int q = 0; q < 10; ++q) {
      cval[(row * 8 + slot) * 10 + q] = bv[q];
      cidx[(row * 8 + slot) * 10 + q] = bi[q];
    }
  }
  __syncthreads();
  if (tid < 16) {
    float fv[10]; int fi[10];
#pragma unroll
    for (int q = 0; q < 10; ++q) { fv[q] = 3.0e38f; fi[q] = 0x7fffffff; }
    for (int s = 0; s < 8; ++s)
      for (int q = 0; q < 10; ++q)
        ins10(fv, fi, cval[(tid * 8 + s) * 10 + q], cidx[(tid * 8 + s) * 10 + q]);
    int* op = idx10 + ((long long)b * N_PNT + n0 + tid) * 12;
    for (int q = 0; q < 10; ++q) op[q] = fi[q];
  }
}

// ---------------- dense edge-conv combine: max_k relu(g*(u[n]+v[nbr]+b)+bt) ----------------
__global__ void combine_dense(const float* __restrict__ U, const float* __restrict__ V,
                              const int* __restrict__ idx10, const float* __restrict__ bias,
                              const float* __restrict__ g, const float* __restrict__ bt,
                              float* __restrict__ dup) {
  int n = blockIdx.x, b = blockIdx.y, o = threadIdx.x;
  float u = U[((long long)b * N_PNT + n) * 256 + o];
  const int* ip = idx10 + ((long long)b * N_PNT + n) * 12;
  float bb = bias[o], gg = g[o], tt = bt[o];
  float m = 0.f;
  for (int k = 0; k < 10; ++k) {
    int mm = ip[k];
    float val = gg * (u + V[((long long)b * N_PNT + mm) * 256 + o] + bb) + tt;
    m = fmaxf(m, val > 0.f ? val : 0.f);
  }
  dup[((long long)b * 256 + o) * N_PNT + n] = m;
}

// ---------------- sparse kNN (n=32, k=2) ----------------
__global__ void sparse_knn(const float* __restrict__ us, int* __restrict__ sknn) {
  __shared__ float xs[256][32];
  int b = blockIdx.x, n = threadIdx.x;
  for (int c = 0; c < 256; ++c) xs[c][n] = us[((long long)b * 256 + c) * N_STK + n];
  __syncthreads();
  float v0 = 3e38f, v1 = 3e38f; int i0 = 0x7fffffff, i1 = 0x7fffffff;
  for (int m = 0; m < 32; ++m) {
    float d = 0.f;
    for (int c = 0; c < 256; ++c) { float t = xs[c][n] - xs[c][m]; d += t * t; }
    if (d < v0 || (d == v0 && m < i0)) { v1 = v0; i1 = i0; v0 = d; i0 = m; }
    else if (d < v1 || (d == v1 && m < i1)) { v1 = d; i1 = m; }
  }
  sknn[(b * 32 + n) * 2 + 0] = i0;
  sknn[(b * 32 + n) * 2 + 1] = i1;
}

// ---------------- sparse edge conv (direct, tiny) ----------------
__global__ void sparse_edge(const float* __restrict__ us, const int* __restrict__ sknn,
                            const float* __restrict__ W, const float* __restrict__ bias,
                            const float* __restrict__ g, const float* __restrict__ bt,
                            float* __restrict__ sup) {
  __shared__ float ctr[256], nb0[256], nb1[256];
  int n = blockIdx.x, b = blockIdx.y, o = threadIdx.x;
  int i0 = sknn[(b * 32 + n) * 2 + 0], i1 = sknn[(b * 32 + n) * 2 + 1];
  const float* ub = us + (long long)b * 256 * N_STK;
  ctr[o] = ub[o * N_STK + n];
  nb0[o] = ub[o * N_STK + i0];
  nb1[o] = ub[o * N_STK + i1];
  __syncthreads();
  float u = 0.f, w0 = 0.f, w1 = 0.f;
  for (int c = 0; c < 256; ++c) {
    float wt = W[c * 256 + o];
    float wb = W[(256 + c) * 256 + o];
    u  += ctr[c] * (wt - wb);
    w0 += nb0[c] * wb;
    w1 += nb1[c] * wb;
  }
  float bb = bias[o], gg = g[o], tt = bt[o];
  float h0 = gg * (u + w0 + bb) + tt; h0 = h0 > 0.f ? h0 : 0.f;
  float h1 = gg * (u + w1 + bb) + tt; h1 = h1 > 0.f ? h1 : 0.f;
  sup[((long long)b * 256 + o) * N_STK + n] = fmaxf(h0, h1);
}

// ---------------- farthest point sampling (one wave per batch) ----------------
__global__ void fps_k(const float* __restrict__ coor, int* __restrict__ fps) {
  __shared__ float cs[32][128];
  int b = blockIdx.x, n = threadIdx.x;
  for (int j = 0; j < 128; ++j) cs[n][j] = coor[((long long)b * 32 + n) * 128 + j];
  __syncthreads();
  float mind = 1e10f;
  int far = 0, myidx = 0;
  for (int it = 0; it < 16; ++it) {
    if (it == n) myidx = far;
    float d = 0.f;
    for (int j = 0; j < 128; ++j) { float t = cs[n][j] - cs[far][j]; d += t * t; }
    mind = fminf(mind, d);
    float best = mind; int bidx = n;
    for (int off = 16; off > 0; off >>= 1) {
      float ov = __shfl_xor(best, off, 32);
      int   oi = __shfl_xor(bidx, off, 32);
      if (ov > best || (ov == best && oi < bidx)) { best = ov; bidx = oi; }
    }
    far = bidx;
  }
  if (n < 16) fps[b * 16 + n] = myidx;
}

// ---------------- output gathers: sp_s and stk_coor_s ----------------
__global__ void gather_out(const float* __restrict__ sup, const float* __restrict__ coor,
                           const int* __restrict__ fps, float* __restrict__ out) {
  int b = blockIdx.x, t = threadIdx.x;
  for (int si = 0; si < 16; ++si) {
    int s = fps[b * 16 + si];
    out[((long long)b * 256 + t) * 16 + si] = sup[((long long)b * 256 + t) * N_STK + s];
    if (t < 128)
      out[OUT_COOR + ((long long)b * 16 + si) * 128 + t] =
          coor[((long long)b * 32 + s) * 128 + t];
  }
}

// ---------------- im2col for downsample conv (stride 2, pad 1) ----------------
__global__ void im2col_ds_k(const float* __restrict__ dup, const int* __restrict__ fps,
                            f16* __restrict__ B) {
  int si = blockIdx.x, b = blockIdx.y, i = threadIdx.x;
  int s = fps[b * 16 + si];
  const float* dp = dup + ((long long)b * 256 + i) * N_PNT + s * N_SPNT;
  f16* bp = B + ((long long)b * 1024 + si * 64) * 768 + i * 3;
  for (int wp = 0; wp < 64; ++wp) {
#pragma unroll
    for (int dw = 0; dw < 3; ++dw) {
      int w = 2 * wp - 1 + dw;
      float v = ((unsigned)w < 128u) ? dp[w] : 0.f;
      bp[(long long)wp * 768 + dw] = (f16)v;
    }
  }
}

// ---------------- f32 -> f16 weight packing ----------------
__global__ void conv_weights(const float* __restrict__ d2sw, const float* __restrict__ dsw,
                             const float* __restrict__ dnW, f16* __restrict__ wd2s,
                             f16* __restrict__ wds, f16* __restrict__ wdn1,
                             f16* __restrict__ wdn2) {
  int idx = blockIdx.x * 256 + threadIdx.x;
  if (idx < 49152)  wd2s[idx] = (f16)d2sw[idx];
  if (idx < 196608) wds[idx]  = (f16)dsw[idx];
  if (idx < 65536) {
    int o = idx >> 8, c = idx & 255;
    float a  = dnW[c * 256 + o];
    float bb = dnW[(256 + c) * 256 + o];
    wdn1[o * 256 + c] = (f16)(a - bb);
    wdn2[o * 256 + c] = (f16)bb;
  }
}

extern "C" void kernel_launch(void* const* d_in, const int* in_sizes, int n_in,
                              void* d_out, int out_size, void* d_ws, size_t ws_size,
                              hipStream_t stream) {
  const float* sparse_fea = (const float*)d_in[0];
  const float* dense_fea  = (const float*)d_in[1];
  const float* stk_coor   = (const float*)d_in[2];
  const float* d2s_w  = (const float*)d_in[3];
  const float* d2s_b  = (const float*)d_in[4];
  const float* d2s_g  = (const float*)d_in[5];
  const float* d2s_bt = (const float*)d_in[6];
  const float* sp_W   = (const float*)d_in[7];
  const float* sp_b   = (const float*)d_in[8];
  const float* sp_g   = (const float*)d_in[9];
  const float* sp_bt  = (const float*)d_in[10];
  const float* dn_W   = (const float*)d_in[11];
  const float* dn_b   = (const float*)d_in[12];
  const float* dn_g   = (const float*)d_in[13];
  const float* dn_bt  = (const float*)d_in[14];
  const float* ds_w   = (const float*)d_in[15];
  const float* ds_b   = (const float*)d_in[16];
  const float* ds_g   = (const float*)d_in[17];
  const float* ds_bt  = (const float*)d_in[18];

  char* ws = (char*)d_ws;
  f16*   XH    = (f16*)(ws + OFF_XH);
  float* SQ    = (float*)(ws + OFF_SQ);
  float* US    = (float*)(ws + OFF_US);
  float* SUP   = (float*)(ws + OFF_SUP);
  int*   FPS   = (int*)(ws + OFF_FPS);
  int*   SKNN  = (int*)(ws + OFF_SKNN);
  int*   IDX10 = (int*)(ws + OFF_IDX10);
  float* DUP   = (float*)(ws + OFF_DUP);
  f16*   WD2S  = (f16*)(ws + OFF_WD2S);
  f16*   WDS   = (f16*)(ws + OFF_WDS);
  f16*   WDN1  = (f16*)(ws + OFF_WDN1);
  f16*   WDN2  = (f16*)(ws + OFF_WDN2);
  f16*   IM2C0 = (f16*)(ws + OFF_BIG0);   // im2col_d2s, later reused for U
  float* Ubuf  = (float*)(ws + OFF_BIG0);
  float* CONV0 = (float*)(ws + OFF_BIG1); // conv_d2s, later reused for im2col_ds
  f16*   IM2DS = (f16*)(ws + OFF_BIG1);
  float* Vbuf  = (float*)(ws + OFF_BIG2);
  float* out   = (float*)d_out;

  // 1. pack weights to f16
  conv_weights<<<768, 256, 0, stream>>>(d2s_w, ds_w, dn_W, WD2S, WDS, WDN1, WDN2);
  // 2. Xh (f16 [b][n][c]) + squared norms
  build_xh<<<dim3(16, 4), 256, 0, stream>>>(dense_fea, sparse_fea, XH, SQ);
  // 3. d2s conv as WMMA GEMM (im2col -> GEMM with fused bias/BN/relu)
  im2col_d2s_k<<<dim3(32, 4), 128, 0, stream>>>(dense_fea, IM2C0);
  gemm_f16<<<dim3(63, 8, 4), 128, 0, stream>>>(WD2S, IM2C0, CONV0, 384, 4032,
      0LL, 4032LL * 384, 128LL * 4032, d2s_g, d2s_b, d2s_bt, 1);
  // 4. max over stroke points + assemble union_sparse
  reduce_union<<<dim3(32, 4), 128, 0, stream>>>(CONV0, sparse_fea, US);
  // 5. dense kNN: WMMA Gram + streaming top-10
  knn_dense<<<dim3(256, 4), 256, 0, stream>>>(XH, SQ, IDX10);
  // 6. edge-conv factored projections U = X(Wt-Wb), V = X Wb (WMMA GEMMs)
  gemm_f16<<<dim3(4, 256, 4), 128, 0, stream>>>(XH, WDN1, Ubuf, 256, 256,
      4096LL * 256, 0LL, 4096LL * 256, nullptr, nullptr, nullptr, 0);
  gemm_f16<<<dim3(4, 256, 4), 128, 0, stream>>>(XH, WDN2, Vbuf, 256, 256,
      4096LL * 256, 0LL, 4096LL * 256, nullptr, nullptr, nullptr, 0);
  // 7. gather/add/BN/relu/max -> dense_up
  combine_dense<<<dim3(4096, 4), 256, 0, stream>>>(Ubuf, Vbuf, IDX10, dn_b, dn_g, dn_bt, DUP);
  // 8. sparse GCN (tiny)
  sparse_knn<<<4, 32, 0, stream>>>(US, SKNN);
  sparse_edge<<<dim3(32, 4), 256, 0, stream>>>(US, SKNN, sp_W, sp_b, sp_g, sp_bt, SUP);
  // 9. FPS + output gathers (sp_s, stk_coor_s)
  fps_k<<<4, 32, 0, stream>>>(stk_coor, FPS);
  gather_out<<<4, 256, 0, stream>>>(SUP, stk_coor, FPS, out);
  // 10. downsample conv as WMMA GEMM with fused BN/relu straight into d_out
  im2col_ds_k<<<dim3(16, 4), 256, 0, stream>>>(DUP, FPS, IM2DS);
  gemm_f16<<<dim3(16, 16, 4), 128, 0, stream>>>(WDS, IM2DS, out + OUT_DN, 768, 1024,
      0LL, 1024LL * 768, 256LL * 1024, ds_g, ds_b, ds_bt, 1);
}
